// BiasedAttention_82557861363858
// MI455X (gfx1250) — matrix operand order repaired and checked by reading, not compiled, not run on hardware
//
#include <hip/hip_runtime.h>
#include <hip/hip_bf16.h>

typedef __attribute__((ext_vector_type(16))) _Float16 v16h;
typedef __attribute__((ext_vector_type(8)))  _Float16 v8h;
typedef __attribute__((ext_vector_type(8)))  float    v8f;
typedef __attribute__((ext_vector_type(4)))  int      v4i;

#define HDIM 256
#define SEQ  2048
#define BATCH 8
#define ATTN_SCALE 0.0625f  // 256^-0.5

// padded LDS strides (halves): rows land on distinct banks for 16-lane reads
#define LD256 264   // 256-wide rows  (528B = 33*16B)
#define LD32  40    // 32-wide rows   (80B  = 5*16B)

#if defined(__AMDGCN__) && __has_builtin(__builtin_amdgcn_global_load_async_to_lds_b128)
#define ASYNC_LDS 1
#else
#define ASYNC_LDS 0
#endif

__device__ inline void g2l16(const _Float16* g, _Float16* l) {
#if ASYNC_LDS
    // param types (from clang diagnostic): (v4i __device__*, v4i __shared__*, imm, imm)
    // generic pointers implicitly convert to HIP target address spaces
    __builtin_amdgcn_global_load_async_to_lds_b128((v4i*)g, (v4i*)l, 0, 0);
#else
    *(v8h*)l = *(const v8h*)g;
#endif
}

__device__ inline void async_fence() {
#if ASYNC_LDS
    asm volatile("s_wait_asynccnt 0" ::: "memory");
#endif
}

__device__ inline v8f v8f_zero() {
    v8f z;
#pragma unroll
    for (int i = 0; i < 8; ++i) z[i] = 0.0f;
    return z;
}

__device__ inline v16h cat8(v8h a, v8h b) {
    return __builtin_shufflevector(a, b, 0,1,2,3,4,5,6,7,8,9,10,11,12,13,14,15);
}

__device__ inline v8f wmma_f16(v16h a, v16h b, v8f c) {
    return __builtin_amdgcn_wmma_f32_16x16x32_f16(
        false, a, false, b, (short)0, c, false, false);
}

// A-fragment (16x32 f16) from row-major LDS tile, stride lda halves.
// lanes 0-15: M=lane, K=0..7 & 16..23 ; lanes 16-31: M=lane-16, K=8..15 & 24..31
__device__ inline v16h load_a_frag(const _Float16* lds, int lda) {
    const int lane = threadIdx.x & 31;
    const _Float16* p = lds + (lane & 15) * lda + ((lane >> 4) * 8);
    v8h x0 = *(const v8h*)(p);
    v8h x1 = *(const v8h*)(p + 16);
    return cat8(x0, x1);
}

// B-fragment (32x16) where logical B[k][n] = T[n][k], T row-major stride ldr.
// lanes 0-15: N=lane, K=0..7 & 16..23 ; lanes 16-31: N=lane-16, K=8..15 & 24..31
__device__ inline v16h load_bt_frag(const _Float16* lds, int ldr) {
    const int lane = threadIdx.x & 31;
    const _Float16* p = lds + (lane & 15) * ldr + ((lane >> 4) * 8);
    v8h x0 = *(const v8h*)(p);
    v8h x1 = *(const v8h*)(p + 16);
    return cat8(x0, x1);
}

// ---------------------------------------------------------------------------
// Kernel 1: out_f16[M,256] = (X_f32 @ W_f32[256,256] + b) * scale
// 256 thr (8 waves), tile 64(M) x 64(N); wave = (wm 0..3, wn 0..1), 2 wmma/step
// ---------------------------------------------------------------------------
__global__ void __launch_bounds__(256) proj_gemm_f32(
    const float* __restrict__ X, const float* __restrict__ W,
    const float* __restrict__ bias, _Float16* __restrict__ out, float scale)
{
    __shared__ _Float16 As[64 * LD32];     // 64 rows x 32 k
    __shared__ _Float16 BsT[64 * LD32];    // transposed: 64 n x 32 k

    const int tid  = threadIdx.x;
    const int lane = tid & 31;
    const int wave = tid >> 5;
    const int wm = wave >> 1, wn = wave & 1;
    const int rowBase = blockIdx.y * 64;
    const int colBase = blockIdx.x * 64;

    v8f acc0 = v8f_zero(), acc1 = v8f_zero();

    for (int kk = 0; kk < HDIM; kk += 32) {
        {   // A tile: 64x32 f32 -> f16, one 16B LDS store per 8 elems
            const int r  = tid >> 2;
            const int cb = (tid & 3) * 8;
            const float* src = X + (size_t)(rowBase + r) * HDIM + kk + cb;
            float4 f0 = *(const float4*)(src);
            float4 f1 = *(const float4*)(src + 4);
            v8h h;
            h[0] = (_Float16)f0.x; h[1] = (_Float16)f0.y;
            h[2] = (_Float16)f0.z; h[3] = (_Float16)f0.w;
            h[4] = (_Float16)f1.x; h[5] = (_Float16)f1.y;
            h[6] = (_Float16)f1.z; h[7] = (_Float16)f1.w;
            *(v8h*)&As[r * LD32 + cb] = h;
        }
        {   // B tile 32k x 64n, stored transposed BsT[n][k]
#pragma unroll
            for (int i2 = 0; i2 < 2; ++i2) {
                const int li = tid + 256 * i2;
                const int k  = li >> 4;
                const int cb = (li & 15) * 4;
                float4 f = *(const float4*)(W + (size_t)(kk + k) * HDIM + colBase + cb);
                BsT[(cb + 0) * LD32 + k] = (_Float16)f.x;
                BsT[(cb + 1) * LD32 + k] = (_Float16)f.y;
                BsT[(cb + 2) * LD32 + k] = (_Float16)f.z;
                BsT[(cb + 3) * LD32 + k] = (_Float16)f.w;
            }
        }
        __syncthreads();
        v16h a  = load_a_frag(&As[wm * 16 * LD32], LD32);
        v16h b0 = load_bt_frag(&BsT[(wn * 32) * LD32], LD32);
        v16h b1 = load_bt_frag(&BsT[(wn * 32 + 16) * LD32], LD32);
        acc0 = wmma_f16(a, b0, acc0);
        acc1 = wmma_f16(a, b1, acc1);
        __syncthreads();
    }

    const int hi = lane >> 4;
    const int ln = lane & 15;
    const int n0 = colBase + wn * 32 + ln;
    const float bn0 = bias[n0];
    const float bn1 = bias[n0 + 16];
#pragma unroll
    for (int r = 0; r < 8; ++r) {
        const int row = rowBase + wm * 16 + r + 8 * hi;
        out[(size_t)row * HDIM + n0]      = (_Float16)((acc0[r] + bn0) * scale);
        out[(size_t)row * HDIM + n0 + 16] = (_Float16)((acc1[r] + bn1) * scale);
    }
}

// ---------------------------------------------------------------------------
// Kernel 2: flash attention, 128 thr (4 waves), 64 q-rows/block, 32-key tiles,
// async double-buffered K/V staging, f32 online softmax, f16 WMMA.
// ---------------------------------------------------------------------------
__global__ void __launch_bounds__(128) flash_attn(
    const _Float16* __restrict__ Qh, const _Float16* __restrict__ Kh,
    const _Float16* __restrict__ Vh, const float* __restrict__ bias,
    _Float16* __restrict__ Oh)
{
    extern __shared__ char smem[];
    _Float16* Qs  = (_Float16*)smem;              // 64 x LD256
    _Float16* Ks0 = Qs  + 64 * LD256;             // 32 x LD256 (row-major keys)
    _Float16* Ks1 = Ks0 + 32 * LD256;
    _Float16* Vt0 = Ks1 + 32 * LD256;             // transposed: 256 h x LD32
    _Float16* Vt1 = Vt0 + 256 * LD32;
    _Float16* Ps  = Vt1 + 256 * LD32;             // 4 waves x 16 x LD32

    const int tid   = threadIdx.x;
    const int lane  = tid & 31;
    const int wave  = tid >> 5;
    const int hi    = lane >> 4;
    const int ln    = lane & 15;
    const int batch = blockIdx.y;
    const int qBase = blockIdx.x * 64;

    const _Float16* Qg = Qh + ((size_t)batch * SEQ + qBase) * HDIM;
    const _Float16* Kg = Kh + (size_t)batch * SEQ * HDIM;
    const _Float16* Vg = Vh + (size_t)batch * SEQ * HDIM;

    // ---- prologue staging: Q (64x256) + K/V tile 0 ----
#pragma unroll
    for (int i = 0; i < 16; ++i) {          // 2048 16B chunks / 128 thr
        const int li = tid + 128 * i;
        const int r  = li >> 5;
        const int c8 = (li & 31) * 8;
        g2l16(Qg + (size_t)r * HDIM + c8, Qs + r * LD256 + c8);
    }
#pragma unroll
    for (int i = 0; i < 8; ++i) {           // K tile 0: 1024 chunks
        const int li = tid + 128 * i;
        const int r  = li >> 5;
        const int c8 = (li & 31) * 8;
        g2l16(Kg + (size_t)r * HDIM + c8, Ks0 + r * LD256 + c8);
    }
#pragma unroll
    for (int i = 0; i < 8; ++i) {           // V tile 0, transposed into Vt0
        const int li  = tid + 128 * i;
        const int key = li >> 5;
        const int h8  = (li & 31) * 8;
        v8h v = *(const v8h*)(Vg + (size_t)key * HDIM + h8);
#pragma unroll
        for (int j = 0; j < 8; ++j) Vt0[(h8 + j) * LD32 + key] = v[j];
    }

    v8f accO[16];
#pragma unroll
    for (int t = 0; t < 16; ++t) accO[t] = v8f_zero();
    float mrow[8], lrow[8];
#pragma unroll
    for (int r = 0; r < 8; ++r) { mrow[r] = -3.0e38f; lrow[r] = 0.0f; }

    async_fence();
    __syncthreads();

    for (int j0 = 0; j0 < SEQ; j0 += 32) {
        const int cur = (j0 >> 5) & 1;
        _Float16* Kcur = cur ? Ks1 : Ks0;
        _Float16* Knxt = cur ? Ks0 : Ks1;
        _Float16* Vcur = cur ? Vt1 : Vt0;
        _Float16* Vnxt = cur ? Vt0 : Vt1;
        const bool haveNext = (j0 + 32) < SEQ;

        // issue async loads for next K tile (lands before end-of-iter fence)
        if (haveNext) {
            const _Float16* Kn = Kg + (size_t)(j0 + 32) * HDIM;
#pragma unroll
            for (int i = 0; i < 8; ++i) {
                const int li = tid + 128 * i;
                const int r  = li >> 5;
                const int c8 = (li & 31) * 8;
                g2l16(Kn + (size_t)r * HDIM + c8, Knxt + r * LD256 + c8);
            }
        }

        // S = Qtile(16x256) * Kcur^T -> two 16x16 f32 tiles
        v8f s0 = v8f_zero(), s1 = v8f_zero();
#pragma unroll
        for (int hh = 0; hh < HDIM; hh += 32) {
            v16h a  = load_a_frag(Qs + wave * 16 * LD256 + hh, LD256);
            v16h b0 = load_bt_frag(Kcur + hh, LD256);              // keys 0..15
            v16h b1 = load_bt_frag(Kcur + 16 * LD256 + hh, LD256); // keys 16..31
            s0 = wmma_f16(a, b0, s0);
            s1 = wmma_f16(a, b1, s1);
        }

        // stage next V tile (global loads overlap the softmax VALU below)
        if (haveNext) {
            const _Float16* Vn = Vg + (size_t)(j0 + 32) * HDIM;
#pragma unroll
            for (int i = 0; i < 8; ++i) {
                const int li  = tid + 128 * i;
                const int key = li >> 5;
                const int h8  = (li & 31) * 8;
                v8h v = *(const v8h*)(Vn + (size_t)key * HDIM + h8);
#pragma unroll
                for (int j = 0; j < 8; ++j) Vnxt[(h8 + j) * LD32 + key] = v[j];
            }
        }

        // + bias (SCALE folded into Q)
        const float* biasRow =
            bias + ((size_t)batch * SEQ + qBase + wave * 16) * SEQ + j0;
        __builtin_prefetch(biasRow + 32, 0, 1);
#pragma unroll
        for (int r = 0; r < 8; ++r) {
            const int m = r + 8 * hi;
            s0[r] += biasRow[(size_t)m * SEQ + ln];
            s1[r] += biasRow[(size_t)m * SEQ + 16 + ln];
        }

        // online softmax (row n-dim lives within each half-wave)
#pragma unroll
        for (int r = 0; r < 8; ++r) {
            float t = fmaxf(s0[r], s1[r]);
            t = fmaxf(t, __shfl_xor(t, 1, 32));
            t = fmaxf(t, __shfl_xor(t, 2, 32));
            t = fmaxf(t, __shfl_xor(t, 4, 32));
            t = fmaxf(t, __shfl_xor(t, 8, 32));
            const float mnew = fmaxf(mrow[r], t);
            const float corr = __expf(mrow[r] - mnew);
            const float p0 = __expf(s0[r] - mnew);
            const float p1 = __expf(s1[r] - mnew);
            float psum = p0 + p1;
            psum += __shfl_xor(psum, 1, 32);
            psum += __shfl_xor(psum, 2, 32);
            psum += __shfl_xor(psum, 4, 32);
            psum += __shfl_xor(psum, 8, 32);
            lrow[r] = lrow[r] * corr + psum;
            mrow[r] = mnew;
#pragma unroll
            for (int t2 = 0; t2 < 16; ++t2) accO[t2][r] *= corr;
            const int m = r + 8 * hi;
            Ps[wave * 16 * LD32 + m * LD32 + ln]      = (_Float16)p0;
            Ps[wave * 16 * LD32 + m * LD32 + 16 + ln] = (_Float16)p1;
        }

        // O += P(16x32) * Vcur(32x256)   (B[k][n] = Vt[n_global][k])
        v16h ap = load_a_frag(Ps + wave * 16 * LD32, LD32);
#pragma unroll
        for (int t2 = 0; t2 < 16; ++t2) {
            v16h bv = load_bt_frag(Vcur + (t2 * 16) * LD32, LD32);
            accO[t2] = wmma_f16(ap, bv, accO[t2]);
        }

        async_fence();      // next K tile landed
        __syncthreads();    // all waves done reading cur buffers / writing next
    }

    // normalize + store f16 O
    _Float16* Og = Oh + ((size_t)batch * SEQ + qBase + wave * 16) * HDIM;
#pragma unroll
    for (int r = 0; r < 8; ++r) {
        const int m = r + 8 * hi;
        const float inv = 1.0f / lrow[r];
#pragma unroll
        for (int t2 = 0; t2 < 16; ++t2)
            Og[(size_t)m * HDIM + t2 * 16 + ln] = (_Float16)(accO[t2][r] * inv);
    }
}

// ---------------------------------------------------------------------------
// Kernel 3: out_f32[M,256] = O_f16 @ W_O[256,256] + b_O
// ---------------------------------------------------------------------------
__global__ void __launch_bounds__(256) out_gemm_f16(
    const _Float16* __restrict__ X, const float* __restrict__ W,
    const float* __restrict__ bias, float* __restrict__ out)
{
    __shared__ _Float16 As[64 * LD32];
    __shared__ _Float16 BsT[64 * LD32];

    const int tid  = threadIdx.x;
    const int lane = tid & 31;
    const int wave = tid >> 5;
    const int wm = wave >> 1, wn = wave & 1;
    const int rowBase = blockIdx.y * 64;
    const int colBase = blockIdx.x * 64;

    v8f acc0 = v8f_zero(), acc1 = v8f_zero();

    for (int kk = 0; kk < HDIM; kk += 32) {
        {   // A tile: direct f16 16B chunks (async path when available)
            const int r  = tid >> 2;
            const int cb = (tid & 3) * 8;
            g2l16(X + (size_t)(rowBase + r) * HDIM + kk + cb, &As[r * LD32 + cb]);
        }
        {   // B tile transposed
#pragma unroll
            for (int i2 = 0; i2 < 2; ++i2) {
                const int li = tid + 256 * i2;
                const int k  = li >> 4;
                const int cb = (li & 15) * 4;
                float4 f = *(const float4*)(W + (size_t)(kk + k) * HDIM + colBase + cb);
                BsT[(cb + 0) * LD32 + k] = (_Float16)f.x;
                BsT[(cb + 1) * LD32 + k] = (_Float16)f.y;
                BsT[(cb + 2) * LD32 + k] = (_Float16)f.z;
                BsT[(cb + 3) * LD32 + k] = (_Float16)f.w;
            }
        }
        async_fence();
        __syncthreads();
        v16h a  = load_a_frag(&As[wm * 16 * LD32], LD32);
        v16h b0 = load_bt_frag(&BsT[(wn * 32) * LD32], LD32);
        v16h b1 = load_bt_frag(&BsT[(wn * 32 + 16) * LD32], LD32);
        acc0 = wmma_f16(a, b0, acc0);
        acc1 = wmma_f16(a, b1, acc1);
        __syncthreads();
    }

    const int hi = lane >> 4;
    const int ln = lane & 15;
    const int n0 = colBase + wn * 32 + ln;
    const float bn0 = bias[n0];
    const float bn1 = bias[n0 + 16];
#pragma unroll
    for (int r = 0; r < 8; ++r) {
        const int row = rowBase + wm * 16 + r + 8 * hi;
        out[(size_t)row * HDIM + n0]      = acc0[r] + bn0;
        out[(size_t)row * HDIM + n0 + 16] = acc1[r] + bn1;
    }
}

// ---------------------------------------------------------------------------
extern "C" void kernel_launch(void* const* d_in, const int* in_sizes, int n_in,
                              void* d_out, int out_size, void* d_ws, size_t ws_size,
                              hipStream_t stream) {
    const float* x         = (const float*)d_in[0];
    const float* attn_bias = (const float*)d_in[1];
    const float* W_Q = (const float*)d_in[2];
    const float* b_Q = (const float*)d_in[3];
    const float* W_K = (const float*)d_in[4];
    const float* b_K = (const float*)d_in[5];
    const float* W_V = (const float*)d_in[6];
    const float* b_V = (const float*)d_in[7];
    const float* W_O = (const float*)d_in[8];
    const float* b_O = (const float*)d_in[9];
    float* out = (float*)d_out;

    const size_t M = (size_t)BATCH * SEQ;          // 16384 rows
    _Float16* Qh = (_Float16*)d_ws;
    _Float16* Kh = Qh + M * HDIM;
    _Float16* Vh = Kh + M * HDIM;
    _Float16* Oh = Vh + M * HDIM;

    dim3 gridP(HDIM / 64, (unsigned)(M / 64));
    proj_gemm_f32<<<gridP, 256, 0, stream>>>(x, W_Q, b_Q, Qh, ATTN_SCALE);
    proj_gemm_f32<<<gridP, 256, 0, stream>>>(x, W_K, b_K, Kh, 1.0f);
    proj_gemm_f32<<<gridP, 256, 0, stream>>>(x, W_V, b_V, Vh, 1.0f);

    const size_t smemBytes =
        (size_t)(64 * LD256 + 2 * 32 * LD256 + 2 * 256 * LD32 + 4 * 16 * LD32)
        * sizeof(_Float16);
    (void)hipFuncSetAttribute((const void*)flash_attn,
                              hipFuncAttributeMaxDynamicSharedMemorySize,
                              (int)smemBytes);
    flash_attn<<<dim3(SEQ / 64, BATCH), 128, smemBytes, stream>>>(
        Qh, Kh, Vh, attn_bias, Oh);

    out_gemm_f16<<<gridP, 256, 0, stream>>>(Oh, W_O, b_O, out);
}
